// SparseAttention_46626164965495
// MI455X (gfx1250) — matrix operand order repaired and checked by reading, not compiled, not run on hardware
//
#include <hip/hip_runtime.h>

// ---------------------------------------------------------------------------
// Problem constants (match reference)
// ---------------------------------------------------------------------------
#define S_LEN 2048
#define DIMX  512
#define NH    8
#define NKV   2
#define DHD   64
#define GQN   4
#define NW    64          // S / CBLK compressed blocks
#define CBLKX 32
#define CDIMX 2048        // CBLK * DH
#define NSELX 8
#define WINX  256
#define ATT_SCALE 0.125f  // 64^-0.5
#define NEGV  -3.4028235e38f
#define NEG10 -3.4028234e37f
#define SWNEG -1.0e30f
#define SPAD  (S_LEN + WINX)   // padded key axis for transposed V

typedef __attribute__((ext_vector_type(16))) _Float16 v16h;
typedef __attribute__((ext_vector_type(8)))  _Float16 v8h;
typedef __attribute__((ext_vector_type(8)))  float    v8f;

// ---------------------------------------------------------------------------
// WMMA helpers (CDNA5 wave32 layouts, 16x16x32 f16 -> f32)
// ---------------------------------------------------------------------------
__device__ __forceinline__ v8f wmma_f16(v16h a, v16h b, v8f c) {
  return __builtin_amdgcn_wmma_f32_16x16x32_f16(false, a, false, b, (short)0, c,
                                                false, false);
}

// A-matrix 16x32 f16, row-major source. lanes 0-15: M=lane, K=0..15
// contiguous; lanes 16-31 K=16..31.
__device__ __forceinline__ v16h load_a16(const _Float16* A, int ld, int row0,
                                         int col0, int lane) {
  int m  = lane & 15;
  int kb = (lane >> 4) << 4;
  const _Float16* p = A + (size_t)(row0 + m) * ld + col0 + kb;
  v8h lo = *(const v8h*)p;
  v8h hi = *(const v8h*)(p + 8);
  v16h r;
#pragma unroll
  for (int i = 0; i < 8; ++i) { r[i] = lo[i]; r[i + 8] = hi[i]; }
  return r;
}

// B-matrix 32x16 from TRANSPOSED (NxK row-major) storage: lane holds column
// n = n0 + lane&15, K halves contiguous; lanes 0-15 K=0..15, 16-31 K=16..31.
__device__ __forceinline__ v16h load_b16_t(const _Float16* Bt, int ldb, int n0,
                                           int k0, int lane, int nmax) {
  int n = n0 + (lane & 15);
  if (n > nmax) n = nmax;
  int kb = (lane >> 4) << 4;
  const _Float16* p = Bt + (size_t)n * ldb + k0 + kb;
  v8h lo = *(const v8h*)p;
  v8h hi = *(const v8h*)(p + 8);
  v16h r;
#pragma unroll
  for (int i = 0; i < 8; ++i) { r[i] = lo[i]; r[i + 8] = hi[i]; }
  return r;
}

// Sliding-window key tile: B^T rows come straight from row-major rk (S x 128).
__device__ __forceinline__ v16h load_b_keys(const _Float16* rkh, int kvh,
                                            int kstart, int j0, int k0,
                                            int lane) {
  int j   = j0 + (lane & 15);
  int row = kstart + j;
  if (row < 0) row = 0;            // masked out later
  int kb = (lane >> 4) << 4;
  const _Float16* p = rkh + (size_t)row * (NKV * DHD) + kvh * DHD + k0 + kb;
  v8h lo = *(const v8h*)p;
  v8h hi = *(const v8h*)(p + 8);
  v16h r;
#pragma unroll
  for (int i = 0; i < 8; ++i) { r[i] = lo[i]; r[i + 8] = hi[i]; }
  return r;
}

// Reductions across a 16-lane half-wave (D-matrix rows live across 16 lanes).
__device__ __forceinline__ float hmax16(float v) {
#pragma unroll
  for (int m = 1; m < 16; m <<= 1) v = fmaxf(v, __shfl_xor(v, m, 32));
  return v;
}
__device__ __forceinline__ float hsum16(float v) {
#pragma unroll
  for (int m = 1; m < 16; m <<= 1) v += __shfl_xor(v, m, 32);
  return v;
}

// ---------------------------------------------------------------------------
// Elementwise / prep kernels
// ---------------------------------------------------------------------------
__global__ void rmsnorm_kernel(const float* __restrict__ x,
                               const float* __restrict__ w,
                               float* __restrict__ xn,
                               _Float16* __restrict__ xnh) {
  __shared__ float red[256];
  int s = blockIdx.x;
  const float* row = x + (size_t)s * DIMX;
  float ss = 0.f;
  for (int i = threadIdx.x; i < DIMX; i += 256) { float v = row[i]; ss += v * v; }
  red[threadIdx.x] = ss;
  __syncthreads();
  for (int st = 128; st > 0; st >>= 1) {
    if ((int)threadIdx.x < st) red[threadIdx.x] += red[threadIdx.x + st];
    __syncthreads();
  }
  float sc = rsqrtf(red[0] * (1.f / DIMX) + 1.1920929e-07f);
  for (int i = threadIdx.x; i < DIMX; i += 256) {
    float v = row[i] * sc * w[i];
    xn[(size_t)s * DIMX + i]  = v;
    xnh[(size_t)s * DIMX + i] = (_Float16)v;
  }
}

__global__ void convT_kernel(const float* __restrict__ W,
                             _Float16* __restrict__ Wt, int K, int N) {
  int i = blockIdx.x * blockDim.x + threadIdx.x;
  if (i >= K * N) return;
  int k = i / N, n = i - k * N;
  Wt[(size_t)n * K + k] = (_Float16)W[i];
}

__global__ void zero_kernel(float* __restrict__ p, int n) {
  int i = blockIdx.x * blockDim.x + threadIdx.x;
  if (i < n) p[i] = 0.f;
}

__global__ void rope_kernel(const float* __restrict__ x, float* __restrict__ o,
                            _Float16* __restrict__ oh, int nheads) {
  int i = blockIdx.x * blockDim.x + threadIdx.x;
  if (i >= S_LEN * nheads * 32) return;
  int pr  = i & 31;
  int t   = i >> 5;
  int hh  = t % nheads;
  int s   = t / nheads;
  float inv = __powf(10000.f, -(float)(2 * pr) * (1.f / 64.f));
  float ang = (float)s * inv;
  float c = __cosf(ang), sn = __sinf(ang);
  size_t base = ((size_t)s * nheads + hh) * DHD;
  float x1 = x[base + 2 * pr], x2 = x[base + 2 * pr + 1];
  float o1 = x1 * c - x2 * sn, o2 = x1 * sn + x2 * c;
  o[base + 2 * pr]      = o1;
  o[base + 2 * pr + 1]  = o2;
  oh[base + 2 * pr]     = (_Float16)o1;
  oh[base + 2 * pr + 1] = (_Float16)o2;
}

// kin[kvh*W+w, c*64+d] = kproj[w*32+c, kvh, d] + k_pos[kvh, c, d]  (f16)
__global__ void prep_comp_in(const float* __restrict__ proj,
                             const float* __restrict__ pos,
                             _Float16* __restrict__ outh) {
  int i = blockIdx.x * blockDim.x + threadIdx.x;
  if (i >= NKV * NW * CBLKX * DHD) return;
  int d   = i & 63;
  int c   = (i >> 6) & 31;
  int w   = (i >> 11) & 63;
  int kvh = i >> 17;
  int s   = w * CBLKX + c;
  float v = proj[(size_t)s * (NKV * DHD) + kvh * DHD + d] +
            pos[((size_t)kvh * CBLKX + c) * DHD + d];
  outh[((size_t)(kvh * NW + w)) * CDIMX + c * DHD + d] = (_Float16)v;
}

// ckf[kvh][0][d]=mem, ckf[kvh][1+j][d]=ck[kvh*64+j][d]  (f16, row-major keys)
__global__ void build_cf_kernel(const float* __restrict__ cc,
                                const float* __restrict__ mem,
                                _Float16* __restrict__ outh) {
  int i = blockIdx.x * blockDim.x + threadIdx.x;
  if (i >= NKV * (NW + 1) * DHD) return;
  int d   = i & 63;
  int j   = (i >> 6) % (NW + 1);
  int kvh = i / ((NW + 1) * DHD);
  float v = (j == 0) ? mem[(size_t)kvh * DHD + d]
                     : cc[((size_t)(kvh * NW + (j - 1))) * DHD + d];
  outh[i] = (_Float16)v;
}

// Transposed compressed values: cvT[kvh][d][j], 96-wide, zero-padded j>=65.
__global__ void cvfT_kernel(const float* __restrict__ cv,
                            const float* __restrict__ mem,
                            _Float16* __restrict__ cvT) {
  int i = blockIdx.x * blockDim.x + threadIdx.x;
  if (i >= NKV * DHD * 96) return;
  int j   = i % 96;
  int t   = i / 96;
  int d   = t & 63;
  int kvh = t >> 6;
  float v = 0.f;
  if (j == 0)       v = mem[(size_t)kvh * DHD + d];
  else if (j <= NW) v = cv[((size_t)(kvh * NW + (j - 1))) * DHD + d];
  cvT[i] = (_Float16)v;
}

// Transposed values with leading zero pad: vT[kvh][d][WINX + s].
__global__ void vpad_trans_kernel(const float* __restrict__ vf,
                                  _Float16* __restrict__ vT) {
  int i = blockIdx.x * blockDim.x + threadIdx.x;
  if (i >= NKV * DHD * SPAD) return;
  int sp  = i % SPAD;
  int t   = i / SPAD;
  int d   = t & 63;
  int kvh = t >> 6;
  float v = (sp < WINX) ? 0.f
                        : vf[(size_t)(sp - WINX) * (NKV * DHD) + kvh * DHD + d];
  vT[i] = (_Float16)v;
}

__global__ void combine_kernel(const float* __restrict__ comb,
                               const float* __restrict__ co,
                               const float* __restrict__ fo,
                               const float* __restrict__ so,
                               _Float16* __restrict__ attnh) {
  int i = blockIdx.x * blockDim.x + threadIdx.x;
  if (i >= S_LEN * NH * DHD) return;
  int hh = (i >> 6) & 7;
  int s  = i >> 9;
  const float* g = comb + (size_t)s * (3 * NH) + hh * 3;
  attnh[i] = (_Float16)(g[0] * co[i] + g[1] * fo[i] + g[2] * so[i]);
}

// ---------------------------------------------------------------------------
// f16 WMMA GEMM: C = act(A(MxK) @ Bt^T(NxK) + bias), f32 out (+f16 copy).
// One wave computes a 16x64 strip: A tile reused across 4 B tiles so each
// k-chunk issues 4 independent WMMAs for 10 b128 loads.  K multiple of 32.
// act: 0 none, 1 relu, 2 sigmoid
// ---------------------------------------------------------------------------
__global__ __launch_bounds__(256) void gemm_wmma_kernel(
    const _Float16* __restrict__ A, int lda, const _Float16* __restrict__ Bt,
    int ldb, const float* __restrict__ bias, float* __restrict__ C, int ldc,
    _Float16* __restrict__ Ch, int M, int N, int K, int act) {
  int lane = threadIdx.x & 31;
  int wave = (blockIdx.x << 3) + (threadIdx.x >> 5);
  int mtiles  = (M + 15) >> 4;
  int nstrips = (N + 63) >> 6;
  if (wave >= mtiles * nstrips) return;
  int mt = wave / nstrips;
  int ns = wave - mt * nstrips;
  int row0 = mt << 4, col0 = ns << 6;
  v8f acc0 = {}, acc1 = {}, acc2 = {}, acc3 = {};
  for (int k0 = 0; k0 < K; k0 += 32) {
    if (k0 + 32 < K) {
      __builtin_prefetch(A + (size_t)(row0 + (lane & 15)) * lda + k0 + 32, 0, 1);
      __builtin_prefetch(Bt + (size_t)(col0 + (lane & 15)) * ldb + k0 + 32, 0, 1);
    }
    v16h a  = load_a16(A, lda, row0, k0, lane);
    v16h b0 = load_b16_t(Bt, ldb, col0,      k0, lane, N - 1);
    v16h b1 = load_b16_t(Bt, ldb, col0 + 16, k0, lane, N - 1);
    v16h b2 = load_b16_t(Bt, ldb, col0 + 32, k0, lane, N - 1);
    v16h b3 = load_b16_t(Bt, ldb, col0 + 48, k0, lane, N - 1);
    acc0 = wmma_f16(a, b0, acc0);
    acc1 = wmma_f16(a, b1, acc1);
    acc2 = wmma_f16(a, b2, acc2);
    acc3 = wmma_f16(a, b3, acc3);
  }
  int nn = lane & 15;
  int mh = (lane >> 4) << 3;
  v8f accs[4] = {acc0, acc1, acc2, acc3};
#pragma unroll
  for (int t = 0; t < 4; ++t) {
    int n = col0 + t * 16 + nn;
    if (n >= N) continue;
    float bv = (bias != nullptr) ? bias[n] : 0.f;
#pragma unroll
    for (int r = 0; r < 8; ++r) {
      int m = row0 + mh + r;
      if (m >= M) continue;
      float x = accs[t][r] + bv;
      if (act == 1)      x = fmaxf(x, 0.f);
      else if (act == 2) x = 1.f / (1.f + __expf(-x));
      C[(size_t)m * ldc + n] = x;
      if (Ch) Ch[(size_t)m * ldc + n] = (_Float16)x;
    }
  }
}

// ---------------------------------------------------------------------------
// Compressed attention (65 keys) + importance scores.  One wave per
// (head, 16-query tile).  Scores & PV via WMMA, softmax in registers.
// ---------------------------------------------------------------------------
__global__ __launch_bounds__(256) void cmp_attn_kernel(
    const _Float16* __restrict__ qh, const _Float16* __restrict__ ckf,
    const _Float16* __restrict__ cvT, float* __restrict__ c_out,
    float* __restrict__ imp) {
  __shared__ _Float16 lp[8][16][96];
  int wid = threadIdx.x >> 5, lane = threadIdx.x & 31;
  int gw = (blockIdx.x << 3) + wid;
  int h  = gw >> 7, st = gw & 127;
  int s0 = st << 4;
  int kvh = h >> 2;
  int nn = lane & 15, mh = (lane >> 4) << 3;
  const _Float16* ckb = ckf + (size_t)kvh * (NW + 1) * DHD;
  const _Float16* cvb = cvT + (size_t)kvh * DHD * 96;
  v16h a0 = load_a16(qh, NH * DHD, s0, h * DHD, lane);
  v16h a1 = load_a16(qh, NH * DHD, s0, h * DHD + 32, lane);
  v8f sc[5];
#pragma unroll
  for (int jt = 0; jt < 5; ++jt) {
    v8f c = {};
    c = wmma_f16(a0, load_b16_t(ckb, DHD, jt * 16, 0, lane, NW), c);
    c = wmma_f16(a1, load_b16_t(ckb, DHD, jt * 16, 32, lane, NW), c);
    sc[jt] = c;
  }
  float mx[8], rs[8];
#pragma unroll
  for (int r = 0; r < 8; ++r) {
    int s = s0 + mh + r;
    float lm = NEGV;
#pragma unroll
    for (int jt = 0; jt < 5; ++jt) {
      int j = jt * 16 + nn;
      float val;
      if (j > NW)                          val = NEGV;     // pad: excluded
      else if (j == 0 || j * CBLKX <= s)   val = sc[jt][r] * ATT_SCALE;
      else                                 val = NEG10;    // reference NEG/10
      if (j >= 1 && j <= NW)
        atomicAdd(&imp[((size_t)kvh * S_LEN + s) * NW + (j - 1)], val * 0.25f);
      sc[jt][r] = val;
      lm = fmaxf(lm, val);
    }
    mx[r] = hmax16(lm);
  }
#pragma unroll
  for (int r = 0; r < 8; ++r) {
    float ls = 0.f;
    int m = mh + r;
#pragma unroll
    for (int jt = 0; jt < 5; ++jt) {
      float p = __expf(sc[jt][r] - mx[r]);
      ls += p;
      lp[wid][m][jt * 16 + nn] = (_Float16)p;
    }
    lp[wid][m][80 + nn] = (_Float16)0.f;   // zero pad K=80..95
    rs[r] = hsum16(ls);
  }
  __syncthreads();
  v8f accO[4] = {};
  for (int kc = 0; kc < 3; ++kc) {
    v16h ap = load_a16(&lp[wid][0][0], 96, 0, kc * 32, lane);
#pragma unroll
    for (int dt = 0; dt < 4; ++dt)
      accO[dt] = wmma_f16(ap, load_b16_t(cvb, 96, dt * 16, kc * 32, lane, 63),
                          accO[dt]);
  }
#pragma unroll
  for (int dt = 0; dt < 4; ++dt)
#pragma unroll
    for (int r = 0; r < 8; ++r)
      c_out[(size_t)(s0 + mh + r) * (NH * DHD) + h * DHD + dt * 16 + nn] =
          accO[dt][r] / rs[r];
}

// ---------------------------------------------------------------------------
// Top-8 block selection per (kvh, s)
// ---------------------------------------------------------------------------
__global__ void topk_kernel(const float* __restrict__ imp,
                            int* __restrict__ selidx,
                            int* __restrict__ selmsk) {
  int idx = blockIdx.x * blockDim.x + threadIdx.x;
  if (idx >= NKV * S_LEN) return;
  const float* row = imp + (size_t)idx * NW;
  unsigned long long used = 0ull;
  for (int t = 0; t < NSELX; ++t) {
    float best = -INFINITY;
    int bi = 0;
    for (int j = 0; j < NW; ++j) {
      if ((used >> j) & 1ull) continue;
      float v = row[j];
      if (v > best) { best = v; bi = j; }
    }
    used |= 1ull << bi;
    selidx[(size_t)idx * NSELX + t] = bi;
    selmsk[(size_t)idx * NSELX + t] = (best > 1e-10f) ? 1 : 0;
  }
}

// ---------------------------------------------------------------------------
// Fine block-sparse attention: one wave per (head, s); lanes cover the 32
// in-block offsets, 9 blocks each (8 selected + diagonal).  Gather-bound ->
// VALU dot products + wave shuffle softmax.
// ---------------------------------------------------------------------------
__global__ __launch_bounds__(128) void fine_attn_kernel(
    const float* __restrict__ rq, const float* __restrict__ rk,
    const float* __restrict__ v, const int* __restrict__ selidx,
    const int* __restrict__ selmsk, float* __restrict__ f_out) {
  __shared__ float pb[4][(NSELX + 1) * CBLKX];
  __shared__ int   ib[4][(NSELX + 1) * CBLKX];
  int wid = threadIdx.x >> 5, lane = threadIdx.x & 31;
  int gw = (blockIdx.x << 2) + wid;
  int h  = gw >> 11;
  int s  = gw & (S_LEN - 1);
  int kvh = h >> 2;
  const float4* qp = (const float4*)(rq + (size_t)s * (NH * DHD) + h * DHD);
  float4 qv[16];
#pragma unroll
  for (int i = 0; i < 16; ++i) qv[i] = qp[i];
  int selbase = (kvh * S_LEN + s) * NSELX;
  float sc[NSELX + 1];
  int   ps[NSELX + 1];
  float mx = NEGV;
#pragma unroll
  for (int t = 0; t <= NSELX; ++t) {
    int b; bool ok;
    if (t < NSELX) { b = selidx[selbase + t]; ok = selmsk[selbase + t] != 0; }
    else           { b = s >> 5;              ok = lane <= (s & 31); }
    int pos = b * CBLKX + lane;
    ps[t] = pos;
    const float4* kp = (const float4*)(rk + (size_t)pos * (NKV * DHD) + kvh * DHD);
    float d = 0.f;
#pragma unroll
    for (int i = 0; i < 16; ++i) {
      float4 kv = kp[i];
      d += qv[i].x * kv.x + qv[i].y * kv.y + qv[i].z * kv.z + qv[i].w * kv.w;
    }
    sc[t] = ok ? d * ATT_SCALE : NEGV;
    mx = fmaxf(mx, sc[t]);
  }
#pragma unroll
  for (int m = 16; m >= 1; m >>= 1) mx = fmaxf(mx, __shfl_xor(mx, m, 32));
  float ls = 0.f;
#pragma unroll
  for (int t = 0; t <= NSELX; ++t) {
    float p = __expf(sc[t] - mx);
    ls += p;
    pb[wid][t * CBLKX + lane] = p;
    ib[wid][t * CBLKX + lane] = ps[t];
  }
#pragma unroll
  for (int m = 16; m >= 1; m >>= 1) ls += __shfl_xor(ls, m, 32);
  __syncthreads();
  float inv = 1.f / ls;
#pragma unroll
  for (int dd = 0; dd < 2; ++dd) {
    int d = lane + dd * 32;
    float acc = 0.f;
    for (int j = 0; j < (NSELX + 1) * CBLKX; ++j)
      acc += pb[wid][j] * v[(size_t)ib[wid][j] * (NKV * DHD) + kvh * DHD + d];
    f_out[(size_t)s * (NH * DHD) + h * DHD + d] = acc * inv;
  }
}

// ---------------------------------------------------------------------------
// Sliding-window attention: flash-style WMMA.  One wave per (head, 16-query
// tile); 512 keys in chunks of 32, online softmax, P transposed through LDS,
// V read from padded transposed layout (contiguous b128 B-loads).
// ---------------------------------------------------------------------------
__global__ __launch_bounds__(256) void swin_attn_kernel(
    const _Float16* __restrict__ rqh, const _Float16* __restrict__ rkh,
    const _Float16* __restrict__ vT, float* __restrict__ s_out) {
  __shared__ _Float16 lp[8][16][32];
  int wid = threadIdx.x >> 5, lane = threadIdx.x & 31;
  int gw = (blockIdx.x << 3) + wid;
  int h  = gw >> 7, st = gw & 127;
  int s0 = st << 4;
  int kvh = h >> 2;
  int kstart = ((s0 >> 8) - 1) * WINX;
  int nn = lane & 15, mh = (lane >> 4) << 3;
  const _Float16* vtb = vT + (size_t)kvh * DHD * SPAD;
  v16h a0 = load_a16(rqh, NH * DHD, s0, h * DHD, lane);
  v16h a1 = load_a16(rqh, NH * DHD, s0, h * DHD + 32, lane);
  float rm[8], rl[8];
  v8f accO[4] = {};
#pragma unroll
  for (int r = 0; r < 8; ++r) { rm[r] = SWNEG; rl[r] = 0.f; }
  for (int jc = 0; jc < 16; ++jc) {
    int jj0 = jc << 5;
    v8f sv0 = {}, sv1 = {};
    sv0 = wmma_f16(a0, load_b_keys(rkh, kvh, kstart, jj0, 0, lane), sv0);
    sv0 = wmma_f16(a1, load_b_keys(rkh, kvh, kstart, jj0, 32, lane), sv0);
    sv1 = wmma_f16(a0, load_b_keys(rkh, kvh, kstart, jj0 + 16, 0, lane), sv1);
    sv1 = wmma_f16(a1, load_b_keys(rkh, kvh, kstart, jj0 + 16, 32, lane), sv1);
#pragma unroll
    for (int r = 0; r < 8; ++r) {
      int qpos = s0 + mh + r;
      int kp0 = kstart + jj0 + nn;
      int kp1 = kp0 + 16;
      float v0 = (kp0 >= 0 && kp0 <= qpos && qpos - kp0 <= WINX)
                     ? sv0[r] * ATT_SCALE : SWNEG;
      float v1 = (kp1 >= 0 && kp1 <= qpos && qpos - kp1 <= WINX)
                     ? sv1[r] * ATT_SCALE : SWNEG;
      float cm = hmax16(fmaxf(v0, v1));
      float mnew  = fmaxf(rm[r], cm);
      float alpha = __expf(rm[r] - mnew);
      rm[r] = mnew;
      float p0 = __expf(v0 - mnew);
      float p1 = __expf(v1 - mnew);
      rl[r] = rl[r] * alpha + hsum16(p0 + p1);
#pragma unroll
      for (int dt = 0; dt < 4; ++dt) accO[dt][r] *= alpha;
      lp[wid][mh + r][nn]      = (_Float16)p0;
      lp[wid][mh + r][nn + 16] = (_Float16)p1;
    }
    __syncthreads();
    v16h ap = load_a16(&lp[wid][0][0], 32, 0, 0, lane);
#pragma unroll
    for (int dt = 0; dt < 4; ++dt) {
      // B from padded transposed V: k index = WINX + kstart + jj0 (+0..31)
      v16h bv = load_b16_t(vtb, SPAD, dt * 16, WINX + kstart + jj0, lane, 63);
      accO[dt] = wmma_f16(ap, bv, accO[dt]);
    }
    __syncthreads();
  }
#pragma unroll
  for (int dt = 0; dt < 4; ++dt)
#pragma unroll
    for (int r = 0; r < 8; ++r)
      s_out[(size_t)(s0 + mh + r) * (NH * DHD) + h * DHD + dt * 16 + nn] =
          accO[dt][r] / rl[r];
}

// ---------------------------------------------------------------------------
// Host launch
// ---------------------------------------------------------------------------
extern "C" void kernel_launch(void* const* d_in, const int* in_sizes, int n_in,
                              void* d_out, int out_size, void* d_ws,
                              size_t ws_size, hipStream_t stream) {
  (void)in_sizes; (void)n_in; (void)out_size; (void)ws_size;
  const float* inp    = (const float*)d_in[0];
  const float* norm_w = (const float*)d_in[1];
  const float* wq     = (const float*)d_in[2];
  const float* wk     = (const float*)d_in[3];
  const float* wv     = (const float*)d_in[4];
  const float* kcw1   = (const float*)d_in[5];
  const float* kcb1   = (const float*)d_in[6];
  const float* kcw2   = (const float*)d_in[7];
  const float* kcb2   = (const float*)d_in[8];
  const float* vcw1   = (const float*)d_in[9];
  const float* vcb1   = (const float*)d_in[10];
  const float* vcw2   = (const float*)d_in[11];
  const float* vcb2   = (const float*)d_in[12];
  const float* mem_kv = (const float*)d_in[13];
  const float* k_pos  = (const float*)d_in[14];
  const float* v_pos  = (const float*)d_in[15];
  const float* comb_w = (const float*)d_in[16];
  const float* comb_b = (const float*)d_in[17];
  const float* wo     = (const float*)d_in[18];
  float* out = (float*)d_out;

  char* base = (char*)d_ws;
  size_t off = 0;
  auto alloc = [&](size_t bytes) -> void* {
    off = (off + 255) & ~(size_t)255;
    void* p = base + off;
    off += bytes;
    return p;
  };
  const size_t SD = (size_t)S_LEN * DIMX;   // 1M
  const size_t SK = (size_t)S_LEN * NKV * DHD;

  float*    xn    = (float*)alloc(SD * 4);
  _Float16* xnh   = (_Float16*)alloc(SD * 2);
  float*    qf    = (float*)alloc(SD * 4);
  _Float16* qh    = (_Float16*)alloc(SD * 2);
  float*    kf    = (float*)alloc(SK * 4);
  float*    vf    = (float*)alloc(SK * 4);
  float*    rqf   = (float*)alloc(SD * 4);
  _Float16* rqh   = (_Float16*)alloc(SD * 2);
  float*    rkf   = (float*)alloc(SK * 4);
  _Float16* rkh   = (_Float16*)alloc(SK * 2);
  _Float16* vT    = (_Float16*)alloc((size_t)NKV * DHD * SPAD * 2);
  _Float16* wqT   = (_Float16*)alloc((size_t)DIMX * DIMX * 2);
  _Float16* wkT   = (_Float16*)alloc((size_t)DIMX * NKV * DHD * 2);
  _Float16* wvT   = (_Float16*)alloc((size_t)DIMX * NKV * DHD * 2);
  _Float16* kcw1T = (_Float16*)alloc((size_t)CDIMX * CDIMX * 2);
  _Float16* kcw2T = (_Float16*)alloc((size_t)CDIMX * DHD * 2);
  _Float16* vcw1T = (_Float16*)alloc((size_t)CDIMX * CDIMX * 2);
  _Float16* vcw2T = (_Float16*)alloc((size_t)CDIMX * DHD * 2);
  _Float16* combT = (_Float16*)alloc((size_t)DIMX * 3 * NH * 2);
  _Float16* woT   = (_Float16*)alloc((size_t)DIMX * DIMX * 2);
  _Float16* kinh  = (_Float16*)alloc((size_t)NKV * NW * CDIMX * 2);
  _Float16* vinh  = (_Float16*)alloc((size_t)NKV * NW * CDIMX * 2);
  float*    khid  = (float*)alloc((size_t)NKV * NW * CDIMX * 4);
  _Float16* khidh = (_Float16*)alloc((size_t)NKV * NW * CDIMX * 2);
  float*    vhid  = (float*)alloc((size_t)NKV * NW * CDIMX * 4);
  _Float16* vhidh = (_Float16*)alloc((size_t)NKV * NW * CDIMX * 2);
  float*    ck32  = (float*)alloc((size_t)NKV * NW * DHD * 4);
  float*    cv32  = (float*)alloc((size_t)NKV * NW * DHD * 4);
  _Float16* ckfh  = (_Float16*)alloc((size_t)NKV * (NW + 1) * DHD * 2);
  _Float16* cvTh  = (_Float16*)alloc((size_t)NKV * DHD * 96 * 2);
  float*    impf  = (float*)alloc((size_t)NKV * S_LEN * NW * 4);
  int*      seli  = (int*)alloc((size_t)NKV * S_LEN * NSELX * 4);
  int*      selm  = (int*)alloc((size_t)NKV * S_LEN * NSELX * 4);
  float*    coutf = (float*)alloc(SD * 4);
  float*    foutf = (float*)alloc(SD * 4);
  float*    soutf = (float*)alloc(SD * 4);
  float*    combf = (float*)alloc((size_t)S_LEN * 3 * NH * 4);
  _Float16* attnh = (_Float16*)alloc(SD * 2);

  auto cdiv = [](int a, int b) { return (a + b - 1) / b; };
  auto gemm = [&](const _Float16* A, int lda, const _Float16* Bt, int ldb,
                  const float* bias, float* C, int ldc, _Float16* Ch, int M,
                  int N, int K, int act) {
    int tiles = cdiv(M, 16) * cdiv(N, 64);
    gemm_wmma_kernel<<<cdiv(tiles, 8), 256, 0, stream>>>(A, lda, Bt, ldb, bias,
                                                         C, ldc, Ch, M, N, K,
                                                         act);
  };
  auto convT = [&](const float* W, _Float16* Wt, int K, int N) {
    convT_kernel<<<cdiv(K * N, 256), 256, 0, stream>>>(W, Wt, K, N);
  };

  // 0) weight conversion (f32 KxN -> f16 NxK) and norm
  rmsnorm_kernel<<<S_LEN, 256, 0, stream>>>(inp, norm_w, xn, xnh);
  convT(wq, wqT, DIMX, DIMX);
  convT(wk, wkT, DIMX, NKV * DHD);
  convT(wv, wvT, DIMX, NKV * DHD);
  convT(kcw1, kcw1T, CDIMX, CDIMX);
  convT(kcw2, kcw2T, CDIMX, DHD);
  convT(vcw1, vcw1T, CDIMX, CDIMX);
  convT(vcw2, vcw2T, CDIMX, DHD);
  convT(comb_w, combT, DIMX, 3 * NH);
  convT(wo, woT, DIMX, DIMX);

  // 1) projections + gates
  gemm(xnh, DIMX, wqT, DIMX, nullptr, qf, DIMX, qh, S_LEN, DIMX, DIMX, 0);
  gemm(xnh, DIMX, wkT, DIMX, nullptr, kf, NKV * DHD, nullptr, S_LEN, NKV * DHD, DIMX, 0);
  gemm(xnh, DIMX, wvT, DIMX, nullptr, vf, NKV * DHD, nullptr, S_LEN, NKV * DHD, DIMX, 0);
  gemm(xnh, DIMX, combT, DIMX, comb_b, combf, 3 * NH, nullptr, S_LEN, 3 * NH, DIMX, 2);

  // 2) rope + transposed/padded V
  rope_kernel<<<cdiv(S_LEN * NH * 32, 256), 256, 0, stream>>>(qf, rqf, rqh, NH);
  rope_kernel<<<cdiv(S_LEN * NKV * 32, 256), 256, 0, stream>>>(kf, rkf, rkh, NKV);
  vpad_trans_kernel<<<cdiv(NKV * DHD * SPAD, 256), 256, 0, stream>>>(vf, vT);

  // 3) compression MLPs
  prep_comp_in<<<cdiv(NKV * NW * CBLKX * DHD, 256), 256, 0, stream>>>(kf, k_pos, kinh);
  prep_comp_in<<<cdiv(NKV * NW * CBLKX * DHD, 256), 256, 0, stream>>>(vf, v_pos, vinh);
  gemm(kinh, CDIMX, kcw1T, CDIMX, kcb1, khid, CDIMX, khidh, NKV * NW, CDIMX, CDIMX, 1);
  gemm(khidh, CDIMX, kcw2T, CDIMX, kcb2, ck32, DHD, nullptr, NKV * NW, DHD, CDIMX, 0);
  gemm(vinh, CDIMX, vcw1T, CDIMX, vcb1, vhid, CDIMX, vhidh, NKV * NW, CDIMX, CDIMX, 1);
  gemm(vhidh, CDIMX, vcw2T, CDIMX, vcb2, cv32, DHD, nullptr, NKV * NW, DHD, CDIMX, 0);
  build_cf_kernel<<<cdiv(NKV * (NW + 1) * DHD, 256), 256, 0, stream>>>(ck32, mem_kv, ckfh);
  cvfT_kernel<<<cdiv(NKV * DHD * 96, 256), 256, 0, stream>>>(cv32, mem_kv + NKV * DHD, cvTh);

  // 4) compressed attention + importance
  zero_kernel<<<cdiv(NKV * S_LEN * NW, 256), 256, 0, stream>>>(impf, NKV * S_LEN * NW);
  cmp_attn_kernel<<<(NH * (S_LEN / 16)) / 8, 256, 0, stream>>>(qh, ckfh, cvTh, coutf, impf);

  // 5) top-k selection + fine attention
  topk_kernel<<<cdiv(NKV * S_LEN, 256), 256, 0, stream>>>(impf, seli, selm);
  fine_attn_kernel<<<(NH * S_LEN) / 4, 128, 0, stream>>>(rqf, rkf, vf, seli, selm, foutf);

  // 6) sliding-window attention
  swin_attn_kernel<<<(NH * (S_LEN / 16)) / 8, 256, 0, stream>>>(rqh, rkh, vT, soutf);

  // 7) combine + output projection
  combine_kernel<<<cdiv(S_LEN * NH * DHD, 256), 256, 0, stream>>>(combf, coutf, foutf, soutf, attnh);
  gemm(attnh, DIMX, woT, DIMX, nullptr, out, DIMX, nullptr, S_LEN, DIMX, DIMX, 0);
}
